// PerceiverResampler_13039520710878
// MI455X (gfx1250) — compile-verified, matrix operand order checked by reading, not run on hardware
//
#include <hip/hip_runtime.h>
#include <cmath>

typedef unsigned short u16;
typedef __attribute__((ext_vector_type(16))) __bf16 v16bf;
typedef __attribute__((ext_vector_type(8)))  float  v8f;

#define B_     8
#define T_     4
#define BT_    32
#define NMED   1024
#define NLAT_  128
#define NKV    1152
#define D_     1024
#define INNER_ 1536
#define H_     16
#define DH_    96
#define DFF_   4096
#define L_     6

__device__ __forceinline__ u16 f2bf(float f) {
  unsigned u = __float_as_uint(f);
  u += 0x7FFFu + ((u >> 16) & 1u);   // round-to-nearest-even
  return (u16)(u >> 16);
}

__device__ __forceinline__ v16bf ld_bf16x16(const u16* p) {
  // 32B operand assembled from two 16B loads (addresses are 16B aligned)
  union { int4 i[2]; v16bf v; } u;
  u.i[0] = ((const int4*)p)[0];
  u.i[1] = ((const int4*)p)[1];
  return u.v;
}

__device__ __forceinline__ v8f wmma_bf16(v16bf a, v16bf b, v8f c) {
  return __builtin_amdgcn_wmma_f32_16x16x32_bf16(false, a, false, b, (short)0, c, false, false);
}

// LDS byte address of a __shared__ object: flat shared addresses carry the
// LDS offset in their low 32 bits.
__device__ __forceinline__ unsigned lds_addr(const void* p) {
  return (unsigned)(unsigned long long)p;
}

// Async 16B global->LDS copy (GVS mode), tracked on ASYNCcnt. Bypasses VGPRs.
__device__ __forceinline__ void async_copy16(unsigned lds_dst, unsigned gbl_off,
                                             const void* sbase) {
  asm volatile("global_load_async_to_lds_b128 %0, %1, %2"
               :
               : "v"(lds_dst), "v"(gbl_off), "s"(sbase)
               : "memory");
}

__device__ __forceinline__ void wait_async0() {
  asm volatile("s_wait_asynccnt 0" ::: "memory");
}

// ---------------------------------------------------------------- small utils
__global__ void bias_kernel(const float* __restrict__ mask, float* __restrict__ bias) {
  int i = blockIdx.x * 256 + threadIdx.x;
  if (i >= B_ * NKV) return;
  int b = i / NKV, j = i % NKV;
  bias[i] = (j < NMED) ? (mask[b * NMED + j] > 0.f ? 0.f : -1e30f) : 0.f;
}

__global__ void lat_init(const float* __restrict__ latents, float* __restrict__ lat) {
  long long i = (long long)blockIdx.x * 256 + threadIdx.x;
  if (i >= (long long)BT_ * NLAT_ * D_) return;
  lat[i] = latents[i % (NLAT_ * D_)];
}

__global__ void conv_kernel(const float* __restrict__ in, u16* __restrict__ out, int n) {
  int i = blockIdx.x * 256 + threadIdx.x;
  if (i < n) out[i] = f2bf(in[i]);
}

// ------------------------------------------------------------------ layernorm
template <bool BF16OUT>
__global__ __launch_bounds__(256)
void ln_kernel(const float* __restrict__ in, const float* __restrict__ g,
               const float* __restrict__ bta, void* __restrict__ out,
               long long chunk_rows, long long chunk_stride, long long out_ld) {
  const long long row = blockIdx.x;
  const int t = threadIdx.x, lane = t & 31, w = t >> 5;
  const float* x = in + row * 1024;
  float xa[4];
  {
    float4 xv = *(const float4*)(x + t * 4);
    xa[0] = xv.x; xa[1] = xv.y; xa[2] = xv.z; xa[3] = xv.w;
  }
  float s  = xa[0] + xa[1] + xa[2] + xa[3];
  float s2 = xa[0]*xa[0] + xa[1]*xa[1] + xa[2]*xa[2] + xa[3]*xa[3];
#pragma unroll
  for (int o = 16; o >= 1; o >>= 1) { s += __shfl_xor(s, o, 32); s2 += __shfl_xor(s2, o, 32); }
  __shared__ float red[16];
  if (lane == 0) { red[w] = s; red[8 + w] = s2; }
  __syncthreads();
  float S = 0.f, S2 = 0.f;
#pragma unroll
  for (int i = 0; i < 8; ++i) { S += red[i]; S2 += red[8 + i]; }
  float mu  = S * (1.0f / 1024.0f);
  float var = S2 * (1.0f / 1024.0f) - mu * mu;
  float inv = rsqrtf(var + 1e-5f);
  long long ob = (row / chunk_rows) * chunk_stride + (row % chunk_rows) * out_ld;
#pragma unroll
  for (int i = 0; i < 4; ++i) {
    int c = t * 4 + i;
    float y = (xa[i] - mu) * inv * g[c] + bta[c];
    if (BF16OUT) ((u16*)out)[ob + c] = f2bf(y);
    else         ((float*)out)[ob + c] = y;
  }
}

// ----------------------------------------------------------------- bf16 GEMM
// C[M,N] = alpha * A[M,K] @ B[K,N] (+ epilogue). Block tile 128x128, k-step 32,
// 8 waves (4x2), each wave: 2x4 grid of 16x16x32 WMMA accumulators.
// A tile: async global->LDS (ASYNCcnt, no VGPR bounce). B tile: transposed into
// LDS on the register path. Next k-tiles prefetched into GL2 (global_prefetch).
// EPI: 0 = f32 store, 1 = bf16 store, 2 = f32 residual add, 3 = GELU -> bf16.
template <int EPI>
__global__ __launch_bounds__(256)
void gemm_bf16(const u16* __restrict__ A, long long acr, long long acs, int lda,
               const u16* __restrict__ Bw, int ldb,
               void* __restrict__ Cv, int ldc, const float* __restrict__ Cres,
               int K, float alpha) {
  __shared__ u16 As[128 * 40];  // [row][k], +8 pad
  __shared__ u16 Bs[128 * 40];  // [n][k] (B^T), +8 pad
  const int t = threadIdx.x;
  const int lane = t & 31, w = t >> 5;
  const int half = lane >> 4, lcol = lane & 15;
  const int wm = w >> 1, wn = w & 1;
  const long long m0 = (long long)blockIdx.y * 128;
  const int n0 = blockIdx.x * 128;

  v8f acc[2][4];
#pragma unroll
  for (int i = 0; i < 2; ++i)
#pragma unroll
    for (int j = 0; j < 4; ++j)
#pragma unroll
      for (int e = 0; e < 8; ++e) acc[i][j][e] = 0.f;

  for (int kb = 0; kb < K; kb += 32) {
    __syncthreads();
    const u16* lastA = A;
    const u16* lastB = Bw;
#pragma unroll
    for (int it = 0; it < 2; ++it) {               // A tile 128x32, async 16B/slot
      int v = t + it * 256;
      int row = v >> 2, kc = (v & 3) * 8;
      long long gr = m0 + row;
      long long eoff = (gr / acr) * acs + (gr % acr) * (long long)lda + kb + kc;
      async_copy16(lds_addr(&As[row * 40 + kc]), (unsigned)(eoff * 2), A);
      lastA = A + eoff;
    }
#pragma unroll
    for (int it = 0; it < 2; ++it) {               // B tile 32x128, transpose in
      int v = t + it * 256;
      int kr = v >> 4, nc = (v & 15) * 8;
      const u16* src = Bw + (long long)(kb + kr) * ldb + n0 + nc;
      int4 d = *(const int4*)src;
      const u16* dp = (const u16*)&d;
#pragma unroll
      for (int i = 0; i < 8; ++i) Bs[(nc + i) * 40 + kr] = dp[i];
      lastB = src;
    }
    if (kb + 32 < K) {                             // pull next k-tiles into GL2
      __builtin_prefetch(lastA + 32, 0, 1);
      __builtin_prefetch(lastB + (long long)32 * ldb, 0, 1);
    }
    wait_async0();
    __syncthreads();
    v16bf av[2], bv[4];
#pragma unroll
    for (int mt = 0; mt < 2; ++mt)
      av[mt] = ld_bf16x16(&As[(wm * 32 + mt * 16 + lcol) * 40 + half * 16]);
#pragma unroll
    for (int nt = 0; nt < 4; ++nt)
      bv[nt] = ld_bf16x16(&Bs[(wn * 64 + nt * 16 + lcol) * 40 + half * 16]);
#pragma unroll
    for (int mt = 0; mt < 2; ++mt)
#pragma unroll
      for (int nt = 0; nt < 4; ++nt)
        acc[mt][nt] = wmma_bf16(av[mt], bv[nt], acc[mt][nt]);
  }

#pragma unroll
  for (int mt = 0; mt < 2; ++mt)
#pragma unroll
    for (int nt = 0; nt < 4; ++nt)
#pragma unroll
      for (int r = 0; r < 8; ++r) {
        long long gm = m0 + wm * 32 + mt * 16 + r + half * 8;
        int gn = n0 + wn * 64 + nt * 16 + lcol;
        long long ci = gm * (long long)ldc + gn;
        float val = acc[mt][nt][r] * alpha;
        if (EPI == 0)      ((float*)Cv)[ci] = val;
        else if (EPI == 1) ((u16*)Cv)[ci] = f2bf(val);
        else if (EPI == 2) ((float*)Cv)[ci] = Cres[ci] + val;
        else {
          float ge = 0.5f * val * (1.0f + erff(val * 0.70710678118654752f));
          ((u16*)Cv)[ci] = f2bf(ge);
        }
      }
}

// ----------------------------------------------------------- flash attention
// One block per (b,t,head). 8 waves x 16 query rows = 128 latents. Streams 9
// KV tiles of 128 tokens through LDS; online softmax; fused P@V. K tile is
// copied with async global->LDS (ASYNCcnt) and overlaps the V transpose; the
// K-tile LDS region is reused as the per-wave P staging area (C->A relayout).
__global__ __launch_bounds__(256)
void flash_attn(const u16* __restrict__ qb, const u16* __restrict__ kvb,
                const float* __restrict__ bias, u16* __restrict__ ob) {
  __shared__ u16 smem[30464];        // 60.9 KB
  u16* vT = smem;                    // [96][136]  V^T tile
  u16* kt = smem + 13056;            // [128][104] K tile (aliases pt)
  u16* pt = smem + 13056;            // 8 waves x [16][136] P staging

  const int t = threadIdx.x, lane = t & 31, w = t >> 5;
  const int half = lane >> 4, lcol = lane & 15;
  const int h = blockIdx.x & 15, bt = blockIdx.x >> 4, b = bt >> 2;
  const long long kvbase = (long long)bt * NKV * (2 * INNER_);

  v16bf qa[3];
  {
    const u16* qp = qb + ((long long)(bt * NLAT_ + w * 16 + lcol)) * INNER_ + h * DH_;
#pragma unroll
    for (int j = 0; j < 3; ++j) qa[j] = ld_bf16x16(qp + j * 32 + half * 16);
  }

  v8f aco[6];
#pragma unroll
  for (int n = 0; n < 6; ++n)
#pragma unroll
    for (int r = 0; r < 8; ++r) aco[n][r] = 0.f;
  float mrun[8], lrun[8];
#pragma unroll
  for (int r = 0; r < 8; ++r) { mrun[r] = -1e30f; lrun[r] = 0.f; }

  for (int tile = 0; tile < 9; ++tile) {
    __syncthreads();                               // prev-iter P/vT reads done
#pragma unroll
    for (int it = 0; it < 6; ++it) {               // K tile 128x96, async copy
      int v = t + it * 256;
      int row = v / 12, c8 = (v % 12) * 8;
      long long eoff = kvbase + (long long)(tile * 128 + row) * (2 * INNER_) + h * DH_ + c8;
      async_copy16(lds_addr(&kt[row * 104 + c8]), (unsigned)(eoff * 2), kvb);
    }
#pragma unroll
    for (int it = 0; it < 6; ++it) {               // V tile transposed to [d][tok]
      int v = t + it * 256;
      int row = v / 12, d8 = (v % 12) * 8;
      const u16* src = kvb + kvbase + (long long)(tile * 128 + row) * (2 * INNER_) + INNER_ + h * DH_ + d8;
      int4 dd = *(const int4*)src;
      const u16* dp = (const u16*)&dd;
#pragma unroll
      for (int i = 0; i < 8; ++i) vT[(d8 + i) * 136 + row] = dp[i];
      if (tile < 8) __builtin_prefetch(src + 128 * (2 * INNER_), 0, 1);
    }
    wait_async0();
    __syncthreads();

    v8f acs[8];                                    // S = Q K^T  (16x128)
#pragma unroll
    for (int nt = 0; nt < 8; ++nt)
#pragma unroll
      for (int r = 0; r < 8; ++r) acs[nt][r] = 0.f;
#pragma unroll
    for (int nt = 0; nt < 8; ++nt)
#pragma unroll
      for (int kk = 0; kk < 3; ++kk) {
        v16bf kv_ = ld_bf16x16(&kt[(nt * 16 + lcol) * 104 + kk * 32 + half * 16]);
        acs[nt] = wmma_bf16(qa[kk], kv_, acs[nt]);
      }
#pragma unroll
    for (int nt = 0; nt < 8; ++nt) {               // column mask bias
      float bv = bias[b * NKV + tile * 128 + nt * 16 + lcol];
#pragma unroll
      for (int r = 0; r < 8; ++r) acs[nt][r] += bv;
    }
    float tm[8], sc[8];
#pragma unroll
    for (int r = 0; r < 8; ++r) {                  // row max (16-lane groups)
      float m = acs[0][r];
#pragma unroll
      for (int nt = 1; nt < 8; ++nt) m = fmaxf(m, acs[nt][r]);
      m = fmaxf(m, __shfl_xor(m, 1, 32));
      m = fmaxf(m, __shfl_xor(m, 2, 32));
      m = fmaxf(m, __shfl_xor(m, 4, 32));
      m = fmaxf(m, __shfl_xor(m, 8, 32));
      tm[r] = m;
    }
#pragma unroll
    for (int r = 0; r < 8; ++r) {
      float nm = fmaxf(mrun[r], tm[r]);
      sc[r] = __expf(mrun[r] - nm);
      mrun[r] = nm;
    }
    float tl[8];
#pragma unroll
    for (int r = 0; r < 8; ++r) tl[r] = 0.f;
#pragma unroll
    for (int nt = 0; nt < 8; ++nt)
#pragma unroll
      for (int r = 0; r < 8; ++r) {
        float p = __expf(acs[nt][r] - mrun[r]);
        acs[nt][r] = p;
        tl[r] += p;
      }
#pragma unroll
    for (int r = 0; r < 8; ++r) {
      float s = tl[r];
      s += __shfl_xor(s, 1, 32); s += __shfl_xor(s, 2, 32);
      s += __shfl_xor(s, 4, 32); s += __shfl_xor(s, 8, 32);
      lrun[r] = lrun[r] * sc[r] + s;
    }
#pragma unroll
    for (int n = 0; n < 6; ++n)
#pragma unroll
      for (int r = 0; r < 8; ++r) aco[n][r] *= sc[r];

    __syncthreads();                               // all waves done reading kt
    u16* pw = pt + w * 16 * 136;                   // C-layout -> LDS (wave-private)
#pragma unroll
    for (int nt = 0; nt < 8; ++nt)
#pragma unroll
      for (int r = 0; r < 8; ++r)
        pw[(r + half * 8) * 136 + nt * 16 + lcol] = f2bf(acs[nt][r]);

#pragma unroll
    for (int kk = 0; kk < 4; ++kk) {               // O += P V
      v16bf ap = ld_bf16x16(&pw[lcol * 136 + kk * 32 + half * 16]);
#pragma unroll
      for (int n = 0; n < 6; ++n) {
        v16bf vv = ld_bf16x16(&vT[(n * 16 + lcol) * 136 + kk * 32 + half * 16]);
        aco[n] = wmma_bf16(ap, vv, aco[n]);
      }
    }
  }

#pragma unroll
  for (int r = 0; r < 8; ++r) {
    float inv = 1.0f / lrun[r];
    long long orow = (long long)(bt * NLAT_ + w * 16 + r + half * 8);
#pragma unroll
    for (int n = 0; n < 6; ++n)
      ob[orow * INNER_ + h * DH_ + n * 16 + lcol] = f2bf(aco[n][r] * inv);
  }
}

// ------------------------------------------------------------------- launch
extern "C" void kernel_launch(void* const* d_in, const int* in_sizes, int n_in,
                              void* d_out, int out_size, void* d_ws, size_t ws_size,
                              hipStream_t stream) {
  const float* x       = (const float*)d_in[0];
  const float* vmask   = (const float*)d_in[1];
  const float* latents = (const float*)d_in[2];
  const float* ln_m_g  = (const float*)d_in[3];
  const float* ln_m_b  = (const float*)d_in[4];
  const float* ln_l_g  = (const float*)d_in[5];
  const float* ln_l_b  = (const float*)d_in[6];
  const float* q_w     = (const float*)d_in[7];
  const float* kv_w    = (const float*)d_in[8];
  const float* out_w   = (const float*)d_in[9];
  const float* ff_ln_g = (const float*)d_in[10];
  const float* ff_ln_b = (const float*)d_in[11];
  const float* ff_w1   = (const float*)d_in[12];
  const float* ff_w2   = (const float*)d_in[13];
  const float* fn_g    = (const float*)d_in[14];
  const float* fn_b    = (const float*)d_in[15];
  float* outp = (float*)d_out;

  // Workspace carve (~420 MB total, 256B-aligned regions).
  char* ws = (char*)d_ws;
  size_t off = 0;
  auto carve = [&](size_t bytes) -> char* {
    char* p = ws + off;
    off = (off + bytes + 255) & ~(size_t)255;
    return p;
  };
  float* bias = (float*)carve((size_t)B_ * NKV * 4);
  float* lat  = (float*)carve((size_t)BT_ * NLAT_ * D_ * 4);
  u16* kvin = (u16*)carve((size_t)BT_ * NKV * D_ * 2);        // [xm; lm] bf16
  u16* qb   = (u16*)carve((size_t)BT_ * NLAT_ * INNER_ * 2);
  u16* kvb  = (u16*)carve((size_t)BT_ * NKV * 2 * INNER_ * 2);
  u16* obuf = (u16*)carve((size_t)BT_ * NLAT_ * INNER_ * 2);
  u16* ffin = (u16*)carve((size_t)BT_ * NLAT_ * D_ * 2);
  u16* hb   = (u16*)carve((size_t)BT_ * NLAT_ * DFF_ * 2);
  u16* wq   = (u16*)carve((size_t)D_ * INNER_ * 2);
  u16* wkv  = (u16*)carve((size_t)D_ * 2 * INNER_ * 2);
  u16* wout = (u16*)carve((size_t)INNER_ * D_ * 2);
  u16* wf1  = (u16*)carve((size_t)D_ * DFF_ * 2);
  u16* wf2  = (u16*)carve((size_t)DFF_ * D_ * 2);
  (void)in_sizes; (void)n_in; (void)out_size; (void)ws_size;

  const long long BIG = 1LL << 40;   // "no chunking" row mapping
  const float qscale = 1.0f / sqrtf(96.0f);

  bias_kernel<<<(B_ * NKV + 255) / 256, 256, 0, stream>>>(vmask, bias);
  lat_init<<<(int)(((long long)BT_ * NLAT_ * D_ + 255) / 256), 256, 0, stream>>>(latents, lat);

  for (int l = 0; l < L_; ++l) {
    conv_kernel<<<(D_ * INNER_ + 255) / 256, 256, 0, stream>>>(q_w + (size_t)l * D_ * INNER_, wq, D_ * INNER_);
    conv_kernel<<<(D_ * 2 * INNER_ + 255) / 256, 256, 0, stream>>>(kv_w + (size_t)l * D_ * 2 * INNER_, wkv, D_ * 2 * INNER_);
    conv_kernel<<<(INNER_ * D_ + 255) / 256, 256, 0, stream>>>(out_w + (size_t)l * INNER_ * D_, wout, INNER_ * D_);
    conv_kernel<<<(D_ * DFF_ + 255) / 256, 256, 0, stream>>>(ff_w1 + (size_t)l * D_ * DFF_, wf1, D_ * DFF_);
    conv_kernel<<<(DFF_ * D_ + 255) / 256, 256, 0, stream>>>(ff_w2 + (size_t)l * DFF_ * D_, wf2, DFF_ * D_);

    // LN(x) -> kvin rows [0,1024) per (b,t); LN(lat) -> kvin rows [1024,1152)
    ln_kernel<true><<<BT_ * NMED, 256, 0, stream>>>(x, ln_m_g + l * D_, ln_m_b + l * D_,
        kvin, (long long)NMED, (long long)NKV * D_, (long long)D_);
    ln_kernel<true><<<BT_ * NLAT_, 256, 0, stream>>>(lat, ln_l_g + l * D_, ln_l_b + l * D_,
        kvin + (size_t)NMED * D_, (long long)NLAT_, (long long)NKV * D_, (long long)D_);

    // Q = scale * lm @ q_w  (A rows are the latent stripes inside kvin)
    gemm_bf16<1><<<dim3(INNER_ / 128, BT_ * NLAT_ / 128), 256, 0, stream>>>(
        kvin + (size_t)NMED * D_, (long long)NLAT_, (long long)NKV * D_, D_,
        wq, INNER_, qb, INNER_, nullptr, D_, qscale);
    // KV = [xm; lm] @ kv_w
    gemm_bf16<1><<<dim3(2 * INNER_ / 128, BT_ * NKV / 128), 256, 0, stream>>>(
        kvin, BIG, 0, D_, wkv, 2 * INNER_, kvb, 2 * INNER_, nullptr, D_, 1.0f);

    flash_attn<<<BT_ * H_, 256, 0, stream>>>(qb, kvb, bias, obuf);

    // lat += O @ out_w
    gemm_bf16<2><<<dim3(D_ / 128, BT_ * NLAT_ / 128), 256, 0, stream>>>(
        obuf, BIG, 0, INNER_, wout, D_, lat, D_, lat, INNER_, 1.0f);

    // FFN: lat += W2 @ gelu(W1 @ LN(lat))
    ln_kernel<true><<<BT_ * NLAT_, 256, 0, stream>>>(lat, ff_ln_g + l * D_, ff_ln_b + l * D_,
        ffin, BIG, 0, (long long)D_);
    gemm_bf16<3><<<dim3(DFF_ / 128, BT_ * NLAT_ / 128), 256, 0, stream>>>(
        ffin, BIG, 0, D_, wf1, DFF_, hb, DFF_, nullptr, D_, 1.0f);
    gemm_bf16<2><<<dim3(D_ / 128, BT_ * NLAT_ / 128), 256, 0, stream>>>(
        hb, BIG, 0, DFF_, wf2, D_, lat, D_, lat, DFF_, 1.0f);
  }

  ln_kernel<false><<<BT_ * NLAT_, 256, 0, stream>>>(lat, fn_g, fn_b, outp, BIG, 0, (long long)D_);
}
